// EnhancedPortfolioGAT_77661598646442
// MI455X (gfx1250) — matrix operand order repaired and checked by reading, not compiled, not run on hardware
//
#include <hip/hip_runtime.h>
#include <cstdint>
#include <cstddef>

typedef __attribute__((ext_vector_type(16))) __bf16 v16bf;
typedef __attribute__((ext_vector_type(8)))  __bf16 v8bf;
typedef __attribute__((ext_vector_type(8)))  float  v8f;

#define NEG_SLOPE 0.2f
#define BN_EPS    1e-5f

__device__ __forceinline__ void atomicMaxF(float* addr, float val) {
  // works for mixed-sign floats with -inf init
  if (val >= 0.0f) atomicMax((int*)addr, __float_as_int(val));
  else             atomicMin((unsigned int*)addr, (unsigned int)__float_as_int(val));
}

// ---------------- prep kernels ----------------

// Wt[n*K + k] = bf16( W[k*Nc + n] )   (W is [K][Nc] row-major)
__global__ void k_transpose_bf16(const float* __restrict__ W, __bf16* __restrict__ Wt,
                                 int K, int Nc) {
  int i = blockIdx.x * blockDim.x + threadIdx.x;
  if (i >= K * Nc) return;
  int n = i / K, k = i - n * K;
  Wt[i] = (__bf16)W[(size_t)k * Nc + n];
}

__global__ void k_bn_to_bf16(const float* __restrict__ x, const float* __restrict__ g,
                             const float* __restrict__ b, const float* __restrict__ m,
                             const float* __restrict__ v, __bf16* __restrict__ out,
                             int total, int C) {
  int i = blockIdx.x * blockDim.x + threadIdx.x;
  if (i >= total) return;
  int c = i % C;
  float t = (x[i] - m[c]) * (g[c] * rsqrtf(v[c] + BN_EPS)) + b[c];
  out[i] = (__bf16)t;
}

__global__ void k_to_bf16(const float* __restrict__ x, __bf16* __restrict__ out, int total) {
  int i = blockIdx.x * blockDim.x + threadIdx.x;
  if (i >= total) return;
  out[i] = (__bf16)x[i];
}

// out = bf16( BN2( ELU( agg + bias ) ) )  , C = 256
__global__ void k_post_layer(const float* __restrict__ agg, const float* __restrict__ bias,
                             const float* __restrict__ g, const float* __restrict__ b,
                             const float* __restrict__ m, const float* __restrict__ v,
                             __bf16* __restrict__ out, int total) {
  int i = blockIdx.x * blockDim.x + threadIdx.x;
  if (i >= total) return;
  int c = i & 255;
  float t = agg[i] + bias[c];
  t = (t > 0.0f) ? t : (expf(t) - 1.0f);                   // ELU
  t = (t - m[c]) * (g[c] * rsqrtf(v[c] + BN_EPS)) + b[c];  // BN
  out[i] = (__bf16)t;
}

__global__ void k_zero_f32(float* __restrict__ p, int total) {
  int i = blockIdx.x * blockDim.x + threadIdx.x;
  if (i < total) p[i] = 0.0f;
}

__global__ void k_init_mx_sm(float* __restrict__ mx, float* __restrict__ sm, int total) {
  int i = blockIdx.x * blockDim.x + threadIdx.x;
  if (i >= total) return;
  mx[i] = -__builtin_inff();
  sm[i] = 0.0f;
}

// ---------------- WMMA GEMM: C[M x ncols] = A[M x K](bf16) * Bt[ncols x K]^T(bf16) ----
// one wave computes a 16x64 output strip (4 x 16x16 tiles): A fragment loaded once
// per k-step and reused across 4 WMMAs (4x less A traffic than 16x16-per-wave).
// A 16x32, B 32x16 in contiguous k order (k = 16*(lane/16) + s, same for both); f32 C/D 16x16 per CDNA5 ISA 7.12.2.
__global__ __launch_bounds__(32) void k_gemm_bf16_wmma(const __bf16* __restrict__ A,
                                                       const __bf16* __restrict__ Bt,
                                                       float* __restrict__ C,
                                                       int K, int ncols) {
  const int lane  = threadIdx.x;          // 0..31
  const int m0    = blockIdx.x * 16;
  const int n0    = blockIdx.y * 64;
  const int r     = lane & 15;
  const int hi16  = lane >> 4;            // 0 or 1
  const int khalfA = hi16 * 16;           // A: upper half-wave holds K+16..K+31
  const int koffB  = hi16 * 16;           // B: upper half-wave holds K+16..K+31

  const __bf16* arow = A + (size_t)(m0 + r) * K;
  const __bf16* brow0 = Bt + (size_t)(n0 + r) * K;
  const size_t  bstride = (size_t)16 * K; // next 16-column tile of Bt

  v8f acc0 = {}, acc1 = {}, acc2 = {}, acc3 = {};
  for (int kb = 0; kb < K; kb += 32) {
    v8bf alo = *(const v8bf*)(arow + kb + khalfA);
    v8bf ahi = *(const v8bf*)(arow + kb + khalfA + 8);
    v16bf a = __builtin_shufflevector(alo, ahi, 0,1,2,3,4,5,6,7,8,9,10,11,12,13,14,15);

    v8bf blo, bhi; v16bf b;
    blo = *(const v8bf*)(brow0 + 0 * bstride + kb + koffB);
    bhi = *(const v8bf*)(brow0 + 0 * bstride + kb + koffB + 8);
    b = __builtin_shufflevector(blo, bhi, 0,1,2,3,4,5,6,7,8,9,10,11,12,13,14,15);
    acc0 = __builtin_amdgcn_wmma_f32_16x16x32_bf16(false, a, false, b, (short)0, acc0, false, false);

    blo = *(const v8bf*)(brow0 + 1 * bstride + kb + koffB);
    bhi = *(const v8bf*)(brow0 + 1 * bstride + kb + koffB + 8);
    b = __builtin_shufflevector(blo, bhi, 0,1,2,3,4,5,6,7,8,9,10,11,12,13,14,15);
    acc1 = __builtin_amdgcn_wmma_f32_16x16x32_bf16(false, a, false, b, (short)0, acc1, false, false);

    blo = *(const v8bf*)(brow0 + 2 * bstride + kb + koffB);
    bhi = *(const v8bf*)(brow0 + 2 * bstride + kb + koffB + 8);
    b = __builtin_shufflevector(blo, bhi, 0,1,2,3,4,5,6,7,8,9,10,11,12,13,14,15);
    acc2 = __builtin_amdgcn_wmma_f32_16x16x32_bf16(false, a, false, b, (short)0, acc2, false, false);

    blo = *(const v8bf*)(brow0 + 3 * bstride + kb + koffB);
    bhi = *(const v8bf*)(brow0 + 3 * bstride + kb + koffB + 8);
    b = __builtin_shufflevector(blo, bhi, 0,1,2,3,4,5,6,7,8,9,10,11,12,13,14,15);
    acc3 = __builtin_amdgcn_wmma_f32_16x16x32_bf16(false, a, false, b, (short)0, acc3, false, false);
  }

  const int mbase = m0 + hi16 * 8;
  float* crow = C + (size_t)mbase * ncols + n0 + r;
#pragma unroll
  for (int i = 0; i < 8; ++i) {
    float* cp = crow + (size_t)i * ncols;
    cp[0]  = acc0[i];
    cp[16] = acc1[i];
    cp[32] = acc2[i];
    cp[48] = acc3[i];
  }
}

// ---------------- attention ----------------

// s_src[n,h] = sum_d H[n, h*32+d]*a_src[h,d] ; same for dst
__global__ void k_scores(const float* __restrict__ H, const float* __restrict__ as,
                         const float* __restrict__ ad, float* __restrict__ ss,
                         float* __restrict__ sd, int N) {
  int i = blockIdx.x * blockDim.x + threadIdx.x;   // n*8 + h
  if (i >= N * 8) return;
  int n = i >> 3, h = i & 7;
  const float* hp = H + (size_t)n * 256 + h * 32;
  const float* ap = as + h * 32;
  const float* bp = ad + h * 32;
  float a = 0.0f, b = 0.0f;
#pragma unroll 8
  for (int d = 0; d < 32; ++d) { float t = hp[d]; a += t * ap[d]; b += t * bp[d]; }
  ss[i] = a; sd[i] = b;
}

__device__ __forceinline__ void edge_sd(const int* __restrict__ ei, int e, int E,
                                        int& s, int& d) {
  if (e < E) { s = ei[e]; d = ei[E + e]; }
  else       { s = e - E; d = e - E; }    // self loops
}

__global__ void k_edge_max(const int* __restrict__ ei, const float* __restrict__ ss,
                           const float* __restrict__ sd, float* __restrict__ mx,
                           int E, int ET) {
  int i = blockIdx.x * blockDim.x + threadIdx.x;
  if (i >= ET * 8) return;
  int e = i >> 3, h = i & 7, s, d;
  edge_sd(ei, e, E, s, d);
  float sc = ss[s * 8 + h] + sd[d * 8 + h];
  sc = (sc >= 0.0f) ? sc : NEG_SLOPE * sc;
  atomicMaxF(&mx[d * 8 + h], sc);
}

__global__ void k_edge_sum(const int* __restrict__ ei, const float* __restrict__ ss,
                           const float* __restrict__ sd, const float* __restrict__ mx,
                           float* __restrict__ sm, int E, int ET) {
  int i = blockIdx.x * blockDim.x + threadIdx.x;
  if (i >= ET * 8) return;
  int e = i >> 3, h = i & 7, s, d;
  edge_sd(ei, e, E, s, d);
  float sc = ss[s * 8 + h] + sd[d * 8 + h];
  sc = (sc >= 0.0f) ? sc : NEG_SLOPE * sc;
  atomicAdd(&sm[d * 8 + h], expf(sc - mx[d * 8 + h]));
}

// one 256-thread block per edge: thread t -> (h = t>>5, d = t&31)
__global__ __launch_bounds__(256) void k_edge_agg(const int* __restrict__ ei,
                                                  const float* __restrict__ H,
                                                  const float* __restrict__ ss,
                                                  const float* __restrict__ sd,
                                                  const float* __restrict__ mx,
                                                  const float* __restrict__ sm,
                                                  float* __restrict__ agg, int E) {
  int e = blockIdx.x;
  int t = threadIdx.x;
  int h = t >> 5, s, d;
  edge_sd(ei, e, E, s, d);
  float sc = ss[s * 8 + h] + sd[d * 8 + h];
  sc = (sc >= 0.0f) ? sc : NEG_SLOPE * sc;
  float alpha = expf(sc - mx[d * 8 + h]) / sm[d * 8 + h];
  atomicAdd(&agg[(size_t)d * 256 + t], H[(size_t)s * 256 + t] * alpha);
}

// ---------------- fused head: BN3(ELU(agg+b2)) + skip, then p1(ReLU), p2 ----------------
__global__ __launch_bounds__(256) void k_head(const float* __restrict__ agg,
                                              const float* __restrict__ b2,
                                              const float* __restrict__ g3,
                                              const float* __restrict__ b3,
                                              const float* __restrict__ m3,
                                              const float* __restrict__ v3,
                                              const float* __restrict__ skipH,
                                              const float* __restrict__ skip_b,
                                              const float* __restrict__ p1W,
                                              const float* __restrict__ p1b,
                                              const float* __restrict__ p2W,
                                              const float* __restrict__ p2b,
                                              float* __restrict__ out, int N) {
  __shared__ float u[8][256];
  const int wave = threadIdx.x >> 5;
  const int lane = threadIdx.x & 31;
  const int n = blockIdx.x * 8 + wave;
  if (n < N) {
    for (int c = lane; c < 256; c += 32) {
      float t = agg[(size_t)n * 256 + c] + b2[c];
      t = (t > 0.0f) ? t : (expf(t) - 1.0f);                       // ELU
      t = (t - m3[c]) * (g3[c] * rsqrtf(v3[c] + BN_EPS)) + b3[c];  // BN3
      t += skipH[(size_t)n * 256 + c] + skip_b[c];                 // skip
      u[wave][c] = t;
    }
  }
  __syncthreads();
  if (n < N) {
    float r = p1b[lane];
#pragma unroll 8
    for (int k = 0; k < 256; ++k) r += u[wave][k] * p1W[k * 32 + lane];
    r = fmaxf(r, 0.0f);
    float part = r * p2W[lane];
#pragma unroll
    for (int off = 16; off; off >>= 1) part += __shfl_xor(part, off, 32);
    if (lane == 0) out[n] = part + p2b[0];
  }
}

// ---------------- launch ----------------
extern "C" void kernel_launch(void* const* d_in, const int* in_sizes, int n_in,
                              void* d_out, int out_size, void* d_ws, size_t ws_size,
                              hipStream_t stream) {
  const float* x      = (const float*)d_in[0];
  const int*   ei     = (const int*)  d_in[1];
  const float* bn1_g  = (const float*)d_in[2];
  const float* bn1_b  = (const float*)d_in[3];
  const float* bn1_m  = (const float*)d_in[4];
  const float* bn1_v  = (const float*)d_in[5];
  const float* W1     = (const float*)d_in[6];
  const float* a1_src = (const float*)d_in[7];
  const float* a1_dst = (const float*)d_in[8];
  const float* b1     = (const float*)d_in[9];
  const float* bn2_g  = (const float*)d_in[10];
  const float* bn2_b  = (const float*)d_in[11];
  const float* bn2_m  = (const float*)d_in[12];
  const float* bn2_v  = (const float*)d_in[13];
  const float* W2     = (const float*)d_in[14];
  const float* a2_src = (const float*)d_in[15];
  const float* a2_dst = (const float*)d_in[16];
  const float* b2     = (const float*)d_in[17];
  const float* bn3_g  = (const float*)d_in[18];
  const float* bn3_b  = (const float*)d_in[19];
  const float* bn3_m  = (const float*)d_in[20];
  const float* bn3_v  = (const float*)d_in[21];
  const float* skip_W = (const float*)d_in[22];
  const float* skip_b = (const float*)d_in[23];
  const float* p1_W   = (const float*)d_in[24];
  const float* p1_b   = (const float*)d_in[25];
  const float* p2_W   = (const float*)d_in[26];
  const float* p2_b   = (const float*)d_in[27];
  float* out = (float*)d_out;

  const int N  = in_sizes[0] / 64;
  const int E  = in_sizes[1] / 2;
  const int ET = E + N;

  // workspace carve-up (256B aligned)
  char* w = (char*)d_ws;
  auto take = [&](size_t bytes) -> char* {
    char* p = w; w += (bytes + 255) & ~(size_t)255; return p;
  };
  float*  H    = (float*) take((size_t)N * 256 * sizeof(float));  // H1 -> H2 -> skipH
  float*  agg  = (float*) take((size_t)N * 256 * sizeof(float));  // agg1 -> agg2
  __bf16* Abuf = (__bf16*)take((size_t)N * 256 * sizeof(__bf16)); // bn1(x) [N,64] -> t2 [N,256]
  __bf16* xraw = (__bf16*)take((size_t)N * 64  * sizeof(__bf16)); // x as bf16 (skip branch)
  float*  ssrc = (float*) take((size_t)N * 8 * sizeof(float));
  float*  sdst = (float*) take((size_t)N * 8 * sizeof(float));
  float*  mx   = (float*) take((size_t)N * 8 * sizeof(float));
  float*  sm   = (float*) take((size_t)N * 8 * sizeof(float));
  __bf16* Wt1  = (__bf16*)take((size_t)256 * 64  * sizeof(__bf16));
  __bf16* Wt2  = (__bf16*)take((size_t)256 * 256 * sizeof(__bf16));
  __bf16* Wts  = (__bf16*)take((size_t)256 * 64  * sizeof(__bf16));

  const dim3 T(256);
  auto cdiv = [](int a, int b) { return (a + b - 1) / b; };
  const int nN64  = N * 64;
  const int nN256 = N * 256;
  const int nN8   = N * 8;
  const int nET8  = ET * 8;
  const dim3 gGemm(cdiv(N, 16), 4);   // 16x64 strips over [N x 256]

  // ---- prep ----
  k_transpose_bf16<<<cdiv(256 * 64, 256), T, 0, stream>>>(W1, Wt1, 64, 256);
  k_transpose_bf16<<<cdiv(256 * 256, 256), T, 0, stream>>>(W2, Wt2, 256, 256);
  k_transpose_bf16<<<cdiv(256 * 64, 256), T, 0, stream>>>(skip_W, Wts, 64, 256);
  k_bn_to_bf16<<<cdiv(nN64, 256), T, 0, stream>>>(x, bn1_g, bn1_b, bn1_m, bn1_v,
                                                  Abuf, nN64, 64);
  k_to_bf16<<<cdiv(nN64, 256), T, 0, stream>>>(x, xraw, nN64);

  // ---- layer 1 ----
  k_gemm_bf16_wmma<<<gGemm, dim3(32), 0, stream>>>(Abuf, Wt1, H, 64, 256);
  k_scores<<<cdiv(nN8, 256), T, 0, stream>>>(H, a1_src, a1_dst, ssrc, sdst, N);
  k_init_mx_sm<<<cdiv(nN8, 256), T, 0, stream>>>(mx, sm, nN8);
  k_zero_f32<<<cdiv(nN256, 256), T, 0, stream>>>(agg, nN256);
  k_edge_max<<<cdiv(nET8, 256), T, 0, stream>>>(ei, ssrc, sdst, mx, E, ET);
  k_edge_sum<<<cdiv(nET8, 256), T, 0, stream>>>(ei, ssrc, sdst, mx, sm, E, ET);
  k_edge_agg<<<dim3(ET), T, 0, stream>>>(ei, H, ssrc, sdst, mx, sm, agg, E);
  k_post_layer<<<cdiv(nN256, 256), T, 0, stream>>>(agg, b1, bn2_g, bn2_b, bn2_m, bn2_v,
                                                   Abuf, nN256);

  // ---- layer 2 ----
  k_gemm_bf16_wmma<<<gGemm, dim3(32), 0, stream>>>(Abuf, Wt2, H, 256, 256);
  k_scores<<<cdiv(nN8, 256), T, 0, stream>>>(H, a2_src, a2_dst, ssrc, sdst, N);
  k_init_mx_sm<<<cdiv(nN8, 256), T, 0, stream>>>(mx, sm, nN8);
  k_zero_f32<<<cdiv(nN256, 256), T, 0, stream>>>(agg, nN256);
  k_edge_max<<<cdiv(nET8, 256), T, 0, stream>>>(ei, ssrc, sdst, mx, E, ET);
  k_edge_sum<<<cdiv(nET8, 256), T, 0, stream>>>(ei, ssrc, sdst, mx, sm, E, ET);
  k_edge_agg<<<dim3(ET), T, 0, stream>>>(ei, H, ssrc, sdst, mx, sm, agg, E);

  // ---- skip GEMM (H now free -> reuse as skipH) ----
  k_gemm_bf16_wmma<<<gGemm, dim3(32), 0, stream>>>(xraw, Wts, H, 64, 256);

  // ---- fused head ----
  k_head<<<cdiv(N, 8), T, 0, stream>>>(agg, b2, bn3_g, bn3_b, bn3_m, bn3_v,
                                       H, skip_b, p1_W, p1_b, p2_W, p2_b, out, N);
}
